// Memory_1580547965106
// MI455X (gfx1250) — compile-verified, matrix operand order checked by reading, not run on hardware
//
#include <hip/hip_runtime.h>
#include <hip/hip_bf16.h>
#include <math.h>

typedef __attribute__((ext_vector_type(16))) _Float16 v16h;
typedef __attribute__((ext_vector_type(8)))  _Float16 v8h;
typedef __attribute__((ext_vector_type(8)))  float    v8f;

#define BATCH  262144
#define FEAT   256
#define NCLS   1000
#define NTILE  63           // ceil(1000/16)
#define INV_T  10.0f        // 1/temperature
#define XMAX   10.0f        // |cos|/T <= ~10 -> fixed softmax max, no running max
#define EPSN   1e-12f

#define TILE_HALVES (16 * FEAT)   // 4096 halves = 8 KB per class tile

// ---------------------------------------------------------------- utilities
__device__ __forceinline__ float block_reduce_256(float v, float* sbuf) {
    int lane = threadIdx.x & 31, wv = threadIdx.x >> 5;
    #pragma unroll
    for (int off = 16; off; off >>= 1) v += __shfl_xor(v, off, 32);
    __syncthreads();                     // protect sbuf across reuses
    if (lane == 0) sbuf[wv] = v;
    __syncthreads();
    float r = sbuf[0];
    #pragma unroll
    for (int i = 1; i < 8; ++i) r += sbuf[i];
    return r;
}

// ------------------------------------------------- 0) zero scratch + loss
__global__ void k_zero(float* sums, float* counts, float* lossp) {
    int i = blockIdx.x * blockDim.x + threadIdx.x;
    if (i < NCLS * FEAT) sums[i] = 0.f;
    if (i < NCLS)        counts[i] = 0.f;
    if (i == 0)          *lossp = 0.f;
}

// ------------------------------------------------- 1) per-row inverse norms
__global__ void k_row_invnorm(const float* __restrict__ feat,
                              float* __restrict__ invnorm) {
    int wave = threadIdx.x >> 5, lane = threadIdx.x & 31;
    int row  = blockIdx.x * 8 + wave;
    const float* p = feat + (size_t)row * FEAT + lane * 8;
    float4 a = *(const float4*)(p);
    float4 b = *(const float4*)(p + 4);
    float s = a.x*a.x + a.y*a.y + a.z*a.z + a.w*a.w
            + b.x*b.x + b.y*b.y + b.z*b.z + b.w*b.w;
    #pragma unroll
    for (int off = 16; off; off >>= 1) s += __shfl_xor(s, off, 32);
    if (lane == 0) invnorm[row] = 1.0f / fmaxf(sqrtf(s), EPSN);
}

// ------------------------------------------------- 2) segment-sum scatter
__global__ void k_scatter(const float* __restrict__ feat,
                          const float* __restrict__ invnorm,
                          const int* __restrict__ label,
                          float* __restrict__ sums,
                          float* __restrict__ counts) {
    int row = blockIdx.x, d = threadIdx.x;
    int c = label[row];
    float v = feat[(size_t)row * FEAT + d] * invnorm[row];
    atomicAdd(&sums[c * FEAT + d], v);      // 1 MB target -> L2 resident
    if (d == 0) atomicAdd(&counts[c], 1.0f);
}

// ------------------------------------------------- 3) center update + EMA
__global__ void k_centers(const float* __restrict__ sums,
                          const float* __restrict__ counts,
                          const float* __restrict__ memory,
                          float* __restrict__ out,          // d_out (loss at [0])
                          _Float16* __restrict__ memh) {
    __shared__ float sbuf[8];
    int c = blockIdx.x, d = threadIdx.x;
    float s    = sums[c * FEAT + d];
    float flag = counts[c] > 0.f ? 1.f : 0.f;

    float ss = block_reduce_256(s * s, sbuf);
    float bc = s * (1.0f / fmaxf(sqrtf(ss), EPSN)) * flag;   // batch_center

    float mem = memory[c * FEAT + d];
    float w = block_reduce_256(mem * bc, sbuf);              // cosine weight
    w = 1.0f - (1.0f - w) * flag;

    float v  = w * mem + (1.0f - w) * bc;
    float vv = block_reduce_256(v * v, sbuf);
    float nm = v * (1.0f / fmaxf(sqrtf(vv), EPSN));          // new_memory row

    out[1 + c * FEAT + d] = nm;                              // f32 output
    memh[c * FEAT + d]    = (_Float16)nm;                    // f16 for WMMA
}

// --------------------------- async Global->LDS staging of one class tile
// LDS layout: [chunk c=0..31][class n=0..15][8 halves]; chunk c covers
// halves [c*8, c*8+8) of a class row. Fragment (kk,half) reads chunks
// kk*4+half (lo) and kk*4+2+half (hi): 16 lanes hit 16 consecutive 16-B
// chunks -> conflict-free ds_load_b128.
__device__ __forceinline__ void issue_tile_copy(const _Float16* __restrict__ memh,
                                                int tile, _Float16* dst /*LDS*/) {
    const int tid = threadIdx.x;
    const int clsBase = tile * 16;
    #pragma unroll
    for (int j = 0; j < 2; ++j) {
        int q = tid + j * 256;                 // 512 chunks of 16 B per tile
        int c = q >> 4, n = q & 15;
        int cls = clsBase + n; if (cls >= NCLS) cls = NCLS - 1;  // pad = dup row
        const _Float16* src = memh + (size_t)cls * FEAT + c * 8;
        unsigned ldsoff = (unsigned)(size_t)(dst + (size_t)q * 8);
        asm volatile("global_load_async_to_lds_b128 %0, %1, off"
                     :: "v"(ldsoff), "v"(src) : "memory");
    }
}

// --------------------------- 4) fused GEMM (WMMA f16) + log-softmax + loss
__global__ void __launch_bounds__(256, 1)
k_gemm_loss(const float* __restrict__ feat,
            const float* __restrict__ invnorm,
            const int*   __restrict__ label,
            const _Float16* __restrict__ memh,
            float* __restrict__ lossp) {
    __shared__ _Float16 sb[2][TILE_HALVES];    // double-buffered B tile (16 KB)

    const int wave = threadIdx.x >> 5, lane = threadIdx.x & 31;
    const int mrow = lane & 15, hh = lane >> 4;
    const int rowTile = (blockIdx.x * 8 + wave) * 16;
    const int row = rowTile + mrow;

    // kick off tile 0 while we build A fragments
    issue_tile_copy(memh, 0, sb[0]);

    const float inv = invnorm[row];

    // A fragments: 16x256 f16 per wave (ISA 16-bit A layout: lane = M,
    // lane-half selects K groups {0-7,16-23} vs {8-15,24-31})
    v16h a[8];
    const float* fr = feat + (size_t)row * FEAT;
    #pragma unroll
    for (int kk = 0; kk < 8; ++kk) {
        const float* p = fr + kk * 32 + hh * 8;
        float4 f0 = *(const float4*)(p);
        float4 f1 = *(const float4*)(p + 4);
        float4 f2 = *(const float4*)(p + 16);
        float4 f3 = *(const float4*)(p + 20);
        v16h t;
        t[0]  = (_Float16)(f0.x * inv); t[1]  = (_Float16)(f0.y * inv);
        t[2]  = (_Float16)(f0.z * inv); t[3]  = (_Float16)(f0.w * inv);
        t[4]  = (_Float16)(f1.x * inv); t[5]  = (_Float16)(f1.y * inv);
        t[6]  = (_Float16)(f1.z * inv); t[7]  = (_Float16)(f1.w * inv);
        t[8]  = (_Float16)(f2.x * inv); t[9]  = (_Float16)(f2.y * inv);
        t[10] = (_Float16)(f2.z * inv); t[11] = (_Float16)(f2.w * inv);
        t[12] = (_Float16)(f3.x * inv); t[13] = (_Float16)(f3.y * inv);
        t[14] = (_Float16)(f3.z * inv); t[15] = (_Float16)(f3.w * inv);
        a[kk] = t;
    }

    int lab[8];
    #pragma unroll
    for (int r = 0; r < 8; ++r) lab[r] = label[rowTile + r + 8 * hh];

    float rs[8], rp[8];                 // fixed-max sum-exp / picked logit
    #pragma unroll
    for (int r = 0; r < 8; ++r) { rs[r] = 0.f; rp[r] = 0.f; }

    // tile 0 staged?
    asm volatile("s_wait_asynccnt 0x0" ::: "memory");
    __syncthreads();

    const int boff = (hh * 16 + mrow) * 8;     // lane's chunk base (halves)

    for (int t = 0; t < NTILE; ++t) {
        const int cur = t & 1;
        if (t + 1 < NTILE)
            issue_tile_copy(memh, t + 1, sb[1 - cur]);   // prefetch next tile

        const _Float16* bt = sb[cur];

        // register double-buffered B fragments: DS loads run one K-step ahead
        v8h lob[2], hib[2];
        lob[0] = *(const v8h*)(bt + boff);
        hib[0] = *(const v8h*)(bt + 32 * 8 + boff);
        v8f c = {};
        #pragma unroll
        for (int kk = 0; kk < 8; ++kk) {
            const int b = kk & 1;
            if (kk + 1 < 8) {
                lob[1 - b] = *(const v8h*)(bt + (kk + 1) * 64 * 8      + boff);
                hib[1 - b] = *(const v8h*)(bt + ((kk + 1) * 64 + 32) * 8 + boff);
            }
            v16h bf = __builtin_shufflevector(lob[b], hib[b],
                        0,1,2,3,4,5,6,7,8,9,10,11,12,13,14,15);
            c = __builtin_amdgcn_wmma_f32_16x16x32_f16(
                    false, a[kk], false, bf, (short)0, c, false, false);
        }

        const int  cls   = t * 16 + mrow;
        const bool valid = cls < NCLS;
        #pragma unroll
        for (int r = 0; r < 8; ++r) {          // |x|<=~10 -> fixed-max softmax
            float x = c[r] * INV_T;
            float e = valid ? __expf(x - XMAX) : 0.0f;
            rs[r] += e;
            if (valid && lab[r] == cls) rp[r] = x;
        }

        asm volatile("s_wait_asynccnt 0x0" ::: "memory");  // my copies landed
        __syncthreads();                                   // all reads done
    }

    // single cross-lane merge: sum-exp and label pick over the 16 class lanes
    float contrib = 0.f;
    #pragma unroll
    for (int r = 0; r < 8; ++r) {
        float s = rs[r], p = rp[r];
        s += __shfl_xor(s, 1, 32);  p += __shfl_xor(p, 1, 32);
        s += __shfl_xor(s, 2, 32);  p += __shfl_xor(p, 2, 32);
        s += __shfl_xor(s, 4, 32);  p += __shfl_xor(p, 4, 32);
        s += __shfl_xor(s, 8, 32);  p += __shfl_xor(p, 8, 32);
        contrib += (XMAX + __logf(s)) - p;     // logsumexp - picked logit
    }
    contrib += __shfl_xor(contrib, 16, 32);    // add other half's 8 rows
    if (lane == 0) atomicAdd(lossp, contrib * (1.0f / (float)BATCH));
}

// ---------------------------------------------------------------- launcher
extern "C" void kernel_launch(void* const* d_in, const int* in_sizes, int n_in,
                              void* d_out, int out_size, void* d_ws, size_t ws_size,
                              hipStream_t stream) {
    (void)in_sizes; (void)n_in; (void)out_size; (void)ws_size;
    const float* feat   = (const float*)d_in[0];
    const int*   label  = (const int*)d_in[1];
    const float* memory = (const float*)d_in[2];
    float* out = (float*)d_out;                      // [0]=loss, [1..]=new_memory

    char* ws = (char*)d_ws;                          // all chunks 16B aligned
    float*    sums    = (float*)(ws);                                      // 1000*256 f32
    float*    counts  = (float*)(ws + (size_t)NCLS * FEAT * 4);            // 1000 f32
    float*    invnorm = (float*)(ws + (size_t)NCLS * FEAT * 4 + NCLS * 4); // 262144 f32
    _Float16* memh    = (_Float16*)(ws + (size_t)NCLS * FEAT * 4 + NCLS * 4
                                       + (size_t)BATCH * 4);               // 1000*256 f16

    k_zero       <<<(NCLS * FEAT + 255) / 256, 256, 0, stream>>>(sums, counts, out);
    k_row_invnorm<<<BATCH / 8,   256, 0, stream>>>(feat, invnorm);
    k_scatter    <<<BATCH,       256, 0, stream>>>(feat, invnorm, label, sums, counts);
    k_centers    <<<NCLS,        256, 0, stream>>>(sums, counts, memory, out, memh);
    k_gemm_loss  <<<BATCH / 128, 256, 0, stream>>>(feat, invnorm, label, memh, out);
}
